// EfficientMixedAttention_57028575756693
// MI455X (gfx1250) — matrix operand order repaired and checked by reading, not compiled, not run on hardware
//
#include <hip/hip_runtime.h>
#include <hip/hip_bf16.h>

// ---------------- problem constants ----------------
#define Bc   64
#define Tc   512
#define Dc   512
#define NTc  77
#define Lc   768
#define Hc   8
#define Nc   (NTc + Tc)   // 589
#define DHc  64

typedef __bf16 bf16;
typedef __attribute__((ext_vector_type(16))) __bf16 v16bf;
typedef __attribute__((ext_vector_type(8)))  __bf16 v8bf;
typedef __attribute__((ext_vector_type(8)))  float  v8f;

static __device__ inline v16bf join16(v8bf lo, v8bf hi) {
  return __builtin_shufflevector(lo, hi, 0,1,2,3,4,5,6,7,8,9,10,11,12,13,14,15);
}

// ---------------- gfx1250 async global->LDS path (guarded) ----------------
#if defined(__has_builtin)
# if __has_builtin(__builtin_amdgcn_global_load_async_to_lds_b128)
#   define HAVE_ASYNC_LDS 1
# endif
#endif
#ifndef HAVE_ASYNC_LDS
# define HAVE_ASYNC_LDS 0
#endif

typedef __attribute__((vector_size(4 * sizeof(int)))) int v4i;
typedef __attribute__((address_space(1))) v4i gv4i;   // global v4i32
typedef __attribute__((address_space(3))) v4i lv4i;   // LDS v4i32

// copy 16 bytes global -> LDS (async on gfx1250 when available)
static __device__ __forceinline__ void g2lds16B(const bf16* g, bf16* l) {
#if HAVE_ASYNC_LDS
  __builtin_amdgcn_global_load_async_to_lds_b128((gv4i*)g, (lv4i*)l, 0, 0);
#else
  *(v8bf*)l = *(const v8bf*)g;
#endif
}

static __device__ __forceinline__ void wait_async_lds() {
#if HAVE_ASYNC_LDS
# if __has_builtin(__builtin_amdgcn_s_wait_asynccnt)
  __builtin_amdgcn_s_wait_asynccnt(0);
# else
  asm volatile("s_wait_asynccnt 0x0" ::: "memory");
# endif
#endif
}

// ---------------- small helpers ----------------
__global__ void f2bf_kernel(const float* __restrict__ in, bf16* __restrict__ out, int n) {
  int i = blockIdx.x * 256 + threadIdx.x;
  if (i < n) out[i] = (bf16)in[i];
}

__global__ void rowcoef_motion_kernel(const float* __restrict__ sm,
                                      float* __restrict__ kadd, float* __restrict__ vscale, int n) {
  int i = blockIdx.x * 256 + threadIdx.x;
  if (i < n) { float m = sm[i]; kadd[i] = (1.0f - m) * -1000000.0f; vscale[i] = m; }
}

__global__ void rowcoef_text_kernel(const int* __restrict__ cond,
                                    float* __restrict__ kadd, float* __restrict__ vscale,
                                    int rowsPerB, int n) {
  int i = blockIdx.x * 256 + threadIdx.x;
  if (i < n) {
    int b = i / rowsPerB;
    int c = cond[b];
    float tc = ((c % 10) > 0) ? 1.0f : 0.0f;
    kadd[i] = (1.0f - tc) * -1000000.0f;
    vscale[i] = tc;
  }
}

// ---------------- layernorm (fp32 in -> bf16 out) ----------------
__global__ __launch_bounds__(256) void ln_kernel(const float* __restrict__ x,
                                                 const float* __restrict__ g,
                                                 const float* __restrict__ bta,
                                                 bf16* __restrict__ out, int width) {
  int row = blockIdx.x;
  const float* xr = x + (size_t)row * width;
  bf16* orow = out + (size_t)row * width;
  float s = 0.f, s2 = 0.f;
  for (int i = threadIdx.x; i < width; i += 256) { float v = xr[i]; s += v; s2 += v * v; }
  for (int o = 16; o; o >>= 1) { s += __shfl_xor(s, o, 32); s2 += __shfl_xor(s2, o, 32); }
  __shared__ float red0[8], red1[8];
  int wid = threadIdx.x >> 5;
  if ((threadIdx.x & 31) == 0) { red0[wid] = s; red1[wid] = s2; }
  __syncthreads();
  s = 0.f; s2 = 0.f;
  #pragma unroll
  for (int i = 0; i < 8; i++) { s += red0[i]; s2 += red1[i]; }
  float mu = s / width;
  float var = s2 / width - mu * mu;
  float r = rsqrtf(var + 1e-5f);
  for (int i = threadIdx.x; i < width; i += 256)
    orow[i] = (bf16)((xr[i] - mu) * r * g[i] + bta[i]);
}

// ---------------- generic projection GEMM (bf16 WMMA, double-buffered, async A) ----------------
// C[M, 512] = A[M,K] @ W[K,512] + bias, epilogue: (c + rowadd[r]) * rowscale[r]
// M must be a multiple of 64; K a multiple of 32. RPB = rows per batch (compile-time).
template<int RPB, bool HASADD, bool HASSCALE>
__global__ __launch_bounds__(256) void gemm_proj_kernel(
    const bf16* __restrict__ A, const bf16* __restrict__ W, const float* __restrict__ bias,
    const float* __restrict__ rowadd, const float* __restrict__ rowscale,
    bf16* __restrict__ out, int M, int K,
    long long outBatchStride, int outRowOffset) {
  __shared__ __align__(16) bf16 sA[2][64][40];     // [buf][m][k]
  __shared__ __align__(16) bf16 sBt[2][128][40];   // [buf][n][k] (transposed W tile)
  const int tid = threadIdx.x, lane = tid & 31, wid = tid >> 5;
  const int m0 = blockIdx.y * 64;
  const int n0 = blockIdx.x * 128;
  const int wm = (wid >> 1) * 16;   // wave M offset (0..48)
  const int wn = (wid & 1) * 64;    // wave N offset (0 or 64)
  v8f zero = {0.f,0.f,0.f,0.f,0.f,0.f,0.f,0.f};
  v8f acc[4] = {zero, zero, zero, zero};

  const int arow = tid >> 2;          // 0..63
  const int acol = (tid & 3) * 8;     // 0,8,16,24
  const int brow = tid >> 3;          // 0..31
  const int bcol = (tid & 7) * 16;    // 0..112
  const int mlane = lane & 15, ksel = lane >> 4;

  const bf16* Ab = A + (size_t)(m0 + arow) * K + acol;

  auto stageA = [&](int k0, int bsel) {
    g2lds16B(Ab + k0, &sA[bsel][arow][acol]);
  };
  auto stageB = [&](int k0, int bsel) {
    const bf16* wp = W + (size_t)(k0 + brow) * 512 + n0 + bcol;
    v8bf w0 = *(const v8bf*)wp;
    v8bf w1 = *(const v8bf*)(wp + 8);
    #pragma unroll
    for (int j = 0; j < 8; j++) sBt[bsel][bcol + j][brow] = w0[j];
    #pragma unroll
    for (int j = 0; j < 8; j++) sBt[bsel][bcol + 8 + j][brow] = w1[j];
  };

  // prologue: stage tile 0
  stageA(0, 0);
  stageB(0, 0);
  wait_async_lds();
  __syncthreads();

  int buf = 0;
  for (int k0 = 0;;) {
    const int nk = k0 + 32;
    if (nk < K) {            // stage next tile into the spare buffer
      stageA(nk, buf ^ 1);
      stageB(nk, buf ^ 1);
    }
    // A fragment: lanes 0-15 K={0..7,16..23}, lanes 16-31 K={8..15,24..31}
    v16bf af = join16(*(const v8bf*)&sA[buf][wm + mlane][ksel * 8],
                      *(const v8bf*)&sA[buf][wm + mlane][16 + ksel * 8]);
    // load all 4 B fragments into distinct registers first, then chain WMMAs
    v16bf bfr[4];
    #pragma unroll
    for (int nt = 0; nt < 4; nt++) {
      int nn = wn + nt * 16 + mlane;
      bfr[nt] = join16(*(const v8bf*)&sBt[buf][nn][ksel * 8],
                       *(const v8bf*)&sBt[buf][nn][16 + ksel * 8]);
    }
    #pragma unroll
    for (int nt = 0; nt < 4; nt++)
      acc[nt] = __builtin_amdgcn_wmma_f32_16x16x32_bf16(false, af, false, bfr[nt],
                                                        (short)0, acc[nt], false, false);
    if (nk >= K) break;
    wait_async_lds();
    __syncthreads();
    buf ^= 1;
    k0 = nk;
  }

  // epilogue: C layout lanes 0-15 -> M=i, lanes 16-31 -> M=8+i
  const int mb = (lane >> 4) * 8;
  #pragma unroll
  for (int nt = 0; nt < 4; nt++) {
    int c = n0 + wn + nt * 16 + mlane;
    float bi = bias[c];
    #pragma unroll
    for (int i = 0; i < 8; i++) {
      int r = m0 + wm + mb + i;
      float v = acc[nt][i] + bi;
      int batch = r / RPB;            // compile-time divisor -> magic multiply
      int rr    = r - batch * RPB;
      if (HASADD)   v += rowadd[r];
      if (HASSCALE) v *= rowscale[r];
      size_t oidx = (size_t)batch * (size_t)outBatchStride
                  + (size_t)(rr + outRowOffset) * 512 + c;
      out[oidx] = (bf16)v;
    }
  }
}

// ---------------- softmax of Q over each 64-wide head chunk (in-place bf16) ----------------
__global__ __launch_bounds__(256) void qsoftmax_kernel(bf16* __restrict__ q) {
  size_t row = blockIdx.x;                 // b*T + t
  int wid = threadIdx.x >> 5, lane = threadIdx.x & 31;
  bf16* p = q + row * 512 + wid * 64 + lane * 2;
  float v0 = (float)p[0], v1 = (float)p[1];
  float mx = fmaxf(v0, v1);
  for (int o = 16; o; o >>= 1) mx = fmaxf(mx, __shfl_xor(mx, o, 32));
  float e0 = __expf(v0 - mx), e1 = __expf(v1 - mx);
  float s = e0 + e1;
  for (int o = 16; o; o >>= 1) s += __shfl_xor(s, o, 32);
  float r = 1.0f / s;
  p[0] = (bf16)(e0 * r);
  p[1] = (bf16)(e1 * r);
}

// ---------------- key softmax stats over N (per (b,d)), single-pass online ----------------
__global__ __launch_bounds__(256) void keystats_kernel(const bf16* __restrict__ key,
                                                       float* __restrict__ kmax,
                                                       float* __restrict__ ksum) {
  int b = blockIdx.y;
  int d = blockIdx.x * 256 + threadIdx.x;   // 0..511
  const bf16* p = key + (size_t)b * Nc * 512 + d;
  float mx = -3.4e38f, s = 0.f;
  for (int n = 0; n < Nc; n++) {
    float v = (float)p[(size_t)n * 512];
    if (v > mx) { s *= __expf(mx - v); mx = v; }
    s += __expf(v - mx);
  }
  kmax[b * 512 + d] = mx;
  ksum[b * 512 + d] = s;
}

// ---------------- attention = softmax(key)^T @ value, per (b,h): 64x64 over N ----------------
__global__ __launch_bounds__(256) void attn_kv_kernel(const bf16* __restrict__ key,
                                                      const bf16* __restrict__ val,
                                                      const float* __restrict__ kmax,
                                                      const float* __restrict__ ksum,
                                                      bf16* __restrict__ att) {
  int b = blockIdx.x >> 3, h = blockIdx.x & 7;
  __shared__ __align__(16) bf16 kT[64][40];   // [d][n]  (normalized key, transposed)
  __shared__ __align__(16) bf16 vT[64][40];   // [l][n]  (value, transposed)
  int tid = threadIdx.x, lane = tid & 31, wid = tid >> 5;
  int wm = (wid >> 1) * 16, wn = (wid & 1) * 32;
  v8f zero = {0.f,0.f,0.f,0.f,0.f,0.f,0.f,0.f};
  v8f acc[2] = {zero, zero};

  int nrow = tid >> 3;            // 0..31
  int dg = (tid & 7) * 8;         // 0..56
  float mx[8], rs[8];
  #pragma unroll
  for (int j = 0; j < 8; j++) {
    int d = h * 64 + dg + j;
    mx[j] = kmax[b * 512 + d];
    rs[j] = 1.0f / ksum[b * 512 + d];
  }
  const size_t base = (size_t)b * Nc * 512 + h * 64;
  const int mlane = lane & 15, ksel = lane >> 4;

  for (int n0 = 0; n0 < Nc; n0 += 32) {
    int n = n0 + nrow;
    if (n < Nc) {
      v8bf kv = *(const v8bf*)(key + base + (size_t)n * 512 + dg);
      v8bf vv = *(const v8bf*)(val + base + (size_t)n * 512 + dg);
      #pragma unroll
      for (int j = 0; j < 8; j++) {
        kT[dg + j][nrow] = (bf16)(__expf((float)kv[j] - mx[j]) * rs[j]);
        vT[dg + j][nrow] = vv[j];
      }
    } else {
      #pragma unroll
      for (int j = 0; j < 8; j++) { kT[dg + j][nrow] = (bf16)0.f; vT[dg + j][nrow] = (bf16)0.f; }
    }
    __syncthreads();
    v16bf af = join16(*(const v8bf*)&kT[wm + mlane][ksel * 8],
                      *(const v8bf*)&kT[wm + mlane][16 + ksel * 8]);
    v16bf bfr[2];
    #pragma unroll
    for (int nt = 0; nt < 2; nt++) {
      int l = wn + nt * 16 + mlane;
      bfr[nt] = join16(*(const v8bf*)&vT[l][ksel * 8],
                       *(const v8bf*)&vT[l][16 + ksel * 8]);
    }
    #pragma unroll
    for (int nt = 0; nt < 2; nt++)
      acc[nt] = __builtin_amdgcn_wmma_f32_16x16x32_bf16(false, af, false, bfr[nt],
                                                        (short)0, acc[nt], false, false);
    __syncthreads();
  }
  const int mb = (lane >> 4) * 8;
  bf16* ap = att + ((size_t)(b * 8 + h)) * 4096;
  #pragma unroll
  for (int nt = 0; nt < 2; nt++) {
    int l = wn + nt * 16 + mlane;
    #pragma unroll
    for (int i = 0; i < 8; i++) {
      int m = wm + mb + i;
      ap[m * 64 + l] = (bf16)acc[nt][i];
    }
  }
}

// ---------------- y = softmax(q) @ attention (+ residual x) ----------------
__global__ __launch_bounds__(256) void final_y_kernel(const bf16* __restrict__ q,
                                                      const bf16* __restrict__ att,
                                                      const float* __restrict__ x,
                                                      float* __restrict__ out) {
  int bh = blockIdx.x;
  int b = bh >> 3, h = bh & 7;
  int t0 = blockIdx.y * 128;
  __shared__ __align__(16) bf16 aT[64][72];   // [l][d] transposed attention tile
  int tid = threadIdx.x, lane = tid & 31, wid = tid >> 5;
  {
    int d = tid >> 2;
    int l0 = (tid & 3) * 16;
    const bf16* ap = att + (size_t)bh * 4096 + d * 64 + l0;
    v8bf a0 = *(const v8bf*)ap;
    v8bf a1 = *(const v8bf*)(ap + 8);
    #pragma unroll
    for (int j = 0; j < 8; j++) aT[l0 + j][d] = a0[j];
    #pragma unroll
    for (int j = 0; j < 8; j++) aT[l0 + 8 + j][d] = a1[j];
  }
  __syncthreads();
  v8f zero = {0.f,0.f,0.f,0.f,0.f,0.f,0.f,0.f};
  v8f acc[4] = {zero, zero, zero, zero};
  const int mlane = lane & 15, ksel = lane >> 4;
  const bf16* qbase = q + ((size_t)b * 512 + t0 + wid * 16 + mlane) * 512 + h * 64;
  #pragma unroll
  for (int k0 = 0; k0 < 64; k0 += 32) {
    v16bf af = join16(*(const v8bf*)(qbase + k0 + ksel * 8),
                      *(const v8bf*)(qbase + k0 + 16 + ksel * 8));
    v16bf bfr[4];
    #pragma unroll
    for (int nt = 0; nt < 4; nt++) {
      int l = nt * 16 + mlane;
      int kb = k0 + ksel * 8;
      bfr[nt] = join16(*(const v8bf*)&aT[l][kb], *(const v8bf*)&aT[l][kb + 16]);
    }
    #pragma unroll
    for (int nt = 0; nt < 4; nt++)
      acc[nt] = __builtin_amdgcn_wmma_f32_16x16x32_bf16(false, af, false, bfr[nt],
                                                        (short)0, acc[nt], false, false);
  }
  const int mb = (lane >> 4) * 8;
  #pragma unroll
  for (int nt = 0; nt < 4; nt++) {
    int c = h * 64 + nt * 16 + mlane;
    #pragma unroll
    for (int i = 0; i < 8; i++) {
      int t = t0 + wid * 16 + mb + i;
      size_t idx = ((size_t)b * 512 + t) * 512 + c;
      out[idx] = x[idx] + acc[nt][i];
    }
  }
}

// ---------------- host-side orchestration ----------------
extern "C" void kernel_launch(void* const* d_in, const int* in_sizes, int n_in,
                              void* d_out, int out_size, void* d_ws, size_t ws_size,
                              hipStream_t stream) {
  const float* x        = (const float*)d_in[0];
  const float* xf       = (const float*)d_in[1];
  const float* src_mask = (const float*)d_in[2];
  const int*   cond     = (const int*)d_in[3];
  const float* norm_g   = (const float*)d_in[4];
  const float* norm_b   = (const float*)d_in[5];
  const float* tnorm_g  = (const float*)d_in[6];
  const float* tnorm_b  = (const float*)d_in[7];
  const float* Wq  = (const float*)d_in[8];
  const float* bq  = (const float*)d_in[9];
  const float* Wkt = (const float*)d_in[10];
  const float* bkt = (const float*)d_in[11];
  const float* Wvt = (const float*)d_in[12];
  const float* bvt = (const float*)d_in[13];
  const float* Wkm = (const float*)d_in[14];
  const float* bkm = (const float*)d_in[15];
  const float* Wvm = (const float*)d_in[16];
  const float* bvm = (const float*)d_in[17];
  float* out = (float*)d_out;

  char* w = (char*)d_ws;
  auto alloc = [&](size_t bytes) { char* p = w; w += (bytes + 255) & ~(size_t)255; return p; };

  bf16* xn   = (bf16*)alloc((size_t)Bc * Tc * Dc * 2);
  bf16* xfn  = (bf16*)alloc((size_t)Bc * NTc * Lc * 2);
  bf16* wq   = (bf16*)alloc((size_t)Dc * Dc * 2);
  bf16* wkm  = (bf16*)alloc((size_t)Dc * Dc * 2);
  bf16* wvm  = (bf16*)alloc((size_t)Dc * Dc * 2);
  bf16* wkt  = (bf16*)alloc((size_t)Lc * Dc * 2);
  bf16* wvt  = (bf16*)alloc((size_t)Lc * Dc * 2);
  bf16* qb   = (bf16*)alloc((size_t)Bc * Tc * Dc * 2);
  bf16* keyb = (bf16*)alloc((size_t)Bc * Nc * Dc * 2);
  bf16* valb = (bf16*)alloc((size_t)Bc * Nc * Dc * 2);
  float* kmax = (float*)alloc((size_t)Bc * Dc * 4);
  float* ksum = (float*)alloc((size_t)Bc * Dc * 4);
  bf16* attb = (bf16*)alloc((size_t)Bc * Hc * DHc * DHc * 2);
  float* kadd_m   = (float*)alloc((size_t)Bc * Tc * 4);
  float* vscale_m = (float*)alloc((size_t)Bc * Tc * 4);
  float* kadd_t   = (float*)alloc((size_t)Bc * NTc * 4);
  float* vscale_t = (float*)alloc((size_t)Bc * NTc * 4);

  // 1. weights -> bf16
  f2bf_kernel<<<(Dc * Dc + 255) / 256, 256, 0, stream>>>(Wq,  wq,  Dc * Dc);
  f2bf_kernel<<<(Dc * Dc + 255) / 256, 256, 0, stream>>>(Wkm, wkm, Dc * Dc);
  f2bf_kernel<<<(Dc * Dc + 255) / 256, 256, 0, stream>>>(Wvm, wvm, Dc * Dc);
  f2bf_kernel<<<(Lc * Dc + 255) / 256, 256, 0, stream>>>(Wkt, wkt, Lc * Dc);
  f2bf_kernel<<<(Lc * Dc + 255) / 256, 256, 0, stream>>>(Wvt, wvt, Lc * Dc);

  // 2. layernorms
  ln_kernel<<<Bc * Tc, 256, 0, stream>>>(x, norm_g, norm_b, xn, Dc);
  ln_kernel<<<Bc * NTc, 256, 0, stream>>>(xf, tnorm_g, tnorm_b, xfn, Lc);

  // 3. per-row mask coefficients
  rowcoef_motion_kernel<<<(Bc * Tc + 255) / 256, 256, 0, stream>>>(src_mask, kadd_m, vscale_m, Bc * Tc);
  rowcoef_text_kernel<<<(Bc * NTc + 255) / 256, 256, 0, stream>>>(cond, kadd_t, vscale_t, NTc, Bc * NTc);

  // 4. projection GEMMs (bf16 WMMA)
  const long long bsK = (long long)Nc * Dc;   // key/value batch stride
  gemm_proj_kernel<Tc, false, false><<<dim3(4, (Bc * Tc) / 64), 256, 0, stream>>>(
      xn, wq, bq, nullptr, nullptr, qb, Bc * Tc, Dc, (long long)Tc * Dc, 0);
  gemm_proj_kernel<Tc, true, false><<<dim3(4, (Bc * Tc) / 64), 256, 0, stream>>>(
      xn, wkm, bkm, kadd_m, nullptr, keyb, Bc * Tc, Dc, bsK, NTc);
  gemm_proj_kernel<Tc, false, true><<<dim3(4, (Bc * Tc) / 64), 256, 0, stream>>>(
      xn, wvm, bvm, nullptr, vscale_m, valb, Bc * Tc, Dc, bsK, NTc);
  gemm_proj_kernel<NTc, true, false><<<dim3(4, (Bc * NTc) / 64), 256, 0, stream>>>(
      xfn, wkt, bkt, kadd_t, nullptr, keyb, Bc * NTc, Lc, bsK, 0);
  gemm_proj_kernel<NTc, false, true><<<dim3(4, (Bc * NTc) / 64), 256, 0, stream>>>(
      xfn, wvt, bvt, nullptr, vscale_t, valb, Bc * NTc, Lc, bsK, 0);

  // 5. softmaxes
  qsoftmax_kernel<<<Bc * Tc, 256, 0, stream>>>(qb);
  keystats_kernel<<<dim3(Dc / 256, Bc), 256, 0, stream>>>(keyb, kmax, ksum);

  // 6. attention = key^T @ value (per b,h) over N
  attn_kv_kernel<<<Bc * Hc, 256, 0, stream>>>(keyb, valb, kmax, ksum, attb);

  // 7. y = q @ attention, out = x + y
  final_y_kernel<<<dim3(Bc * Hc, Tc / 128), 256, 0, stream>>>(qb, attb, x, out);
}
